// RelativePositionSelfAttention_65687229825826
// MI455X (gfx1250) — matrix operand order repaired and checked by reading, not compiled, not run on hardware
//
#include <hip/hip_runtime.h>
#include <hip/hip_bf16.h>

// Problem sizes (fixed by the reference)
#define BSZ   16
#define LTOK  1024
#define DIMC  768
#define NH    12
#define HD    64
#define NQKV  2304   // 3*DIM
#define MTOT  16384  // B*L

typedef __attribute__((ext_vector_type(16))) __bf16 v16bf;
typedef __attribute__((ext_vector_type(8)))  __bf16 bf16x8;
typedef __attribute__((ext_vector_type(4)))  __bf16 bf16x4;
typedef __attribute__((ext_vector_type(8)))  float  v8f;

union AFrag { v16bf v; bf16x8 h[2]; };

__device__ __forceinline__ __bf16 f2bf(float f) {
    unsigned u = __builtin_bit_cast(unsigned, f);
    u += 0x7FFFu + ((u >> 16) & 1u);           // round-to-nearest-even
    unsigned short s = (unsigned short)(u >> 16);
    return __builtin_bit_cast(__bf16, s);
}

__device__ __forceinline__ v8f wmma_bf16(const AFrag& a, const AFrag& b, v8f c) {
    // D = A(16x32 bf16) x B(32x16 bf16) + C(16x16 f32)
    return __builtin_amdgcn_wmma_f32_16x16x32_bf16(
        false, a.v, false, b.v, (short)0, c, false, false);
}

// LDS byte address (wave-relative) of a __shared__ object, for async-copy VDST.
__device__ __forceinline__ unsigned lds_addr_of(const void* p) {
    return (unsigned)(uintptr_t)(__attribute__((address_space(3))) const void*)p;
}

// Async DMA: 16 bytes per lane, global -> LDS.  Tracked by ASYNCcnt.
__device__ __forceinline__ void async_copy_b128(unsigned lds_off, const void* gptr) {
    asm volatile("global_load_async_to_lds_b128 %0, %1, off"
                 :: "v"(lds_off), "v"((unsigned long long)(uintptr_t)gptr)
                 : "memory");
}

template<int N>
__device__ __forceinline__ void wait_asynccnt() {
    asm volatile("s_wait_asynccnt %0" :: "i"(N) : "memory");
}

// ---------------------------------------------------------------------------
// Kernel 1: QKV projection GEMM  [16384,768] x [768,2304] (+b_qkv)
// f32 inputs are converted to bf16 while staging tiles (async copy cannot
// convert, so this kernel keeps register-side staging).
// Epilogue scatters:  Q*(1/8) -> q_ws[B,H,L,64] bf16
//                     K       -> k_ws[B,H,L,64] bf16
//                     V       -> vt_ws[B,H,64,L] bf16  (transposed for PV WMMA)
// ---------------------------------------------------------------------------
#define BM 128
#define BN 64
#define BK 32
#define AP (BK + 8)   // LDS pitch (halves): 80B rows -> 16B-aligned frag loads
#define BP (BK + 8)

__global__ __launch_bounds__(256)
void qkv_gemm_kernel(const float* __restrict__ A,      // [16384,768]
                     const float* __restrict__ W,      // [768,2304]
                     const float* __restrict__ bvec,   // [2304]
                     __bf16* __restrict__ q_ws,
                     __bf16* __restrict__ k_ws,
                     __bf16* __restrict__ vt_ws) {
    __shared__ __bf16 As [BM][AP];   // row-major MxK tile
    __shared__ __bf16 BsT[BN][BP];   // transposed: BsT[n][k]

    const int tid  = threadIdx.x;
    const int lane = tid & 31;
    const int wave = tid >> 5;       // 0..7
    const int wm   = wave & 3;       // 4 M-quadrants of 32 rows
    const int wn   = wave >> 2;      // 2 N-halves of 32 cols
    const int hi   = (lane >= 16);
    const int l16  = lane & 15;

    const int mBlock = blockIdx.y * BM;
    const int nBlock = blockIdx.x * BN;

    v8f acc[2][2];
    v8f zero = {};
    acc[0][0] = zero; acc[0][1] = zero; acc[1][0] = zero; acc[1][1] = zero;

    for (int k0 = 0; k0 < DIMC; k0 += BK) {
        // Stage A tile (128x32 f32 -> bf16): 256 thr * 4 rows * float4
        {
            const int c4 = (tid & 7) * 4;
            const int r0 = tid >> 3;
#pragma unroll
            for (int rr = 0; rr < 4; ++rr) {
                const int r = r0 + rr * 32;
                const float4 f = *reinterpret_cast<const float4*>(
                    &A[(size_t)(mBlock + r) * DIMC + k0 + c4]);
                bf16x4 b4 = { f2bf(f.x), f2bf(f.y), f2bf(f.z), f2bf(f.w) };
                *reinterpret_cast<bf16x4*>(&As[r][c4]) = b4;
            }
            if (k0 + BK < DIMC)   // speculative prefetch of next K-step A lines
                __builtin_prefetch(&A[(size_t)(mBlock + r0) * DIMC + k0 + BK], 0, 3);
        }
        // Stage B tile transposed (32x64 f32 -> BsT[64][32] bf16)
        {
            const int n4 = (tid & 15) * 4;
            const int kr = tid >> 4;
#pragma unroll
            for (int kk = 0; kk < 2; ++kk) {
                const int k = kr + kk * 16;
                const float4 f = *reinterpret_cast<const float4*>(
                    &W[(size_t)(k0 + k) * NQKV + nBlock + n4]);
                BsT[n4 + 0][k] = f2bf(f.x);
                BsT[n4 + 1][k] = f2bf(f.y);
                BsT[n4 + 2][k] = f2bf(f.z);
                BsT[n4 + 3][k] = f2bf(f.w);
            }
        }
        __syncthreads();
#pragma unroll
        for (int mt = 0; mt < 2; ++mt) {
            AFrag a;
            const int row = wm * 32 + mt * 16 + l16;
            a.h[0] = *reinterpret_cast<const bf16x8*>(&As[row][hi ? 8 : 0]);
            a.h[1] = *reinterpret_cast<const bf16x8*>(&As[row][16 + (hi ? 8 : 0)]);
#pragma unroll
            for (int nt = 0; nt < 2; ++nt) {
                AFrag b;
                const int col = wn * 32 + nt * 16 + l16;
                b.h[0] = *reinterpret_cast<const bf16x8*>(&BsT[col][hi ? 8 : 0]);
                b.h[1] = *reinterpret_cast<const bf16x8*>(&BsT[col][16 + (hi ? 8 : 0)]);
                acc[mt][nt] = wmma_bf16(a, b, acc[mt][nt]);
            }
        }
        __syncthreads();
    }

    // Epilogue: add bias, scatter into Q/K/Vt workspaces
#pragma unroll
    for (int mt = 0; mt < 2; ++mt) {
#pragma unroll
        for (int nt = 0; nt < 2; ++nt) {
#pragma unroll
            for (int r = 0; r < 8; ++r) {
                const int gr = mBlock + wm * 32 + mt * 16 + (hi ? r + 8 : r);
                const int gc = nBlock + wn * 32 + nt * 16 + l16;
                float val = acc[mt][nt][r] + bvec[gc];
                const int part = gc / DIMC;       // 0=Q 1=K 2=V (tile never straddles)
                const int c    = gc % DIMC;
                const int h    = c >> 6;
                const int hd   = c & 63;
                const int bi   = gr >> 10;
                const int lt   = gr & 1023;
                const size_t bh = (size_t)bi * NH + h;
                if (part == 0)
                    q_ws[(bh * LTOK + lt) * HD + hd] = f2bf(val * 0.125f);
                else if (part == 1)
                    k_ws[(bh * LTOK + lt) * HD + hd] = f2bf(val);
                else
                    vt_ws[(bh * HD + hd) * LTOK + lt] = f2bf(val);
            }
        }
    }
}

// ---------------------------------------------------------------------------
// Kernel 2: flash attention.  One wave per 16-row q-tile; the 4 waves of a
// block share (b,h), so K/V 32-token chunks are DMA'd once per block into LDS
// with GLOBAL_LOAD_ASYNC_TO_LDS_B128 (ASYNCcnt), double-buffered so chunk i+1
// streams in while chunk i is in the matrix pipe.  256 WMMAs / wave.
// ---------------------------------------------------------------------------
__global__ __launch_bounds__(128)
void attn_kernel(const __bf16* __restrict__ q_ws,     // [B,H,L,64]
                 const __bf16* __restrict__ k_ws,     // [B,H,L,64]
                 const __bf16* __restrict__ vt_ws,    // [B,H,64,L]
                 const float*  __restrict__ bias_table, // [3969,12]
                 const int*    __restrict__ rel_index,  // [1024,1024]
                 __bf16* __restrict__ xout) {           // [B,L,768] bf16
    __shared__ __bf16 Ks[2][32][64];     // token-major K chunk  (2 x 4 KB)
    __shared__ __bf16 Vs[2][64][32];     // d-major V chunk      (2 x 4 KB)
    __shared__ __bf16 Pbuf[4][16][40];   // per-wave P staging   (5 KB)

    const int tid  = threadIdx.x;
    const int lane = tid & 31;
    const int wave = tid >> 5;
    const int hi   = (lane >= 16);
    const int l16  = lane & 15;

    const int w  = blockIdx.x * 4 + wave;   // 4 | 64 => block-uniform (b,h)
    const int qt = w & 63;                  // q-tile (64 per head)
    const int h  = (w >> 6) % NH;
    const int b  = w / (64 * NH);
    const size_t bh = (size_t)b * NH + h;

    const __bf16* Qp = q_ws + bh * LTOK * HD;
    const __bf16* Kp = k_ws + bh * LTOK * HD;
    const __bf16* Vt = vt_ws + bh * HD * LTOK;

    // Q fragments for rows qt*16..+15 (Q already scaled by 1/8)
    const int qrow = qt * 16 + l16;
    AFrag qf[2];
    qf[0].h[0] = *reinterpret_cast<const bf16x8*>(&Qp[qrow * HD + (hi ? 8 : 0)]);
    qf[0].h[1] = *reinterpret_cast<const bf16x8*>(&Qp[qrow * HD + 16 + (hi ? 8 : 0)]);
    qf[1].h[0] = *reinterpret_cast<const bf16x8*>(&Qp[qrow * HD + 32 + (hi ? 8 : 0)]);
    qf[1].h[1] = *reinterpret_cast<const bf16x8*>(&Qp[qrow * HD + 48 + (hi ? 8 : 0)]);

    // Async-copy lane assignments (128 threads move 4 KB per buffer)
    const int krow = tid >> 2, kseg = (tid & 3) * 16;   // K: 32 rows x 4 segs
    const int vrow = tid >> 1, vseg = (tid & 1) * 16;   // V: 64 rows x 2 segs
    const unsigned ksLds[2] = { lds_addr_of(&Ks[0][krow][kseg]),
                                lds_addr_of(&Ks[1][krow][kseg]) };
    const unsigned vsLds[2] = { lds_addr_of(&Vs[0][vrow][vseg]),
                                lds_addr_of(&Vs[1][vrow][vseg]) };

    v8f O[4];
    v8f zero = {};
    O[0] = zero; O[1] = zero; O[2] = zero; O[3] = zero;
    float mrow[8], lsum[8];
#pragma unroll
    for (int r = 0; r < 8; ++r) { mrow[r] = -3.0e38f; lsum[r] = 0.0f; }

    // Pipeline prologue: DMA chunk 0 into buffer 0
    async_copy_b128(ksLds[0], &Kp[(size_t)krow * HD + kseg]);
    async_copy_b128(vsLds[0], &Vt[(size_t)vrow * LTOK + vseg]);

    for (int i = 0; i < LTOK / 32; ++i) {
        const int ck  = i * 32;
        const int cur = i & 1;
        if (i + 1 < LTOK / 32) {
            // Stream next chunk into the other buffer while we compute this one
            const int nxt = cur ^ 1;
            async_copy_b128(ksLds[nxt], &Kp[(size_t)(ck + 32 + krow) * HD + kseg]);
            async_copy_b128(vsLds[nxt], &Vt[(size_t)vrow * LTOK + ck + 32 + vseg]);
            wait_asynccnt<2>();   // 2 in-flight = chunk i+1; chunk i has landed
        } else {
            wait_asynccnt<0>();
        }
        __syncthreads();          // chunk i's LDS tiles visible to all waves

        v8f S[2]; S[0] = zero; S[1] = zero;
#pragma unroll
        for (int c = 0; c < 2; ++c) {
            const int tok = c * 16 + l16;
            AFrag kb;
            kb.h[0] = *reinterpret_cast<const bf16x8*>(&Ks[cur][tok][hi ? 8 : 0]);
            kb.h[1] = *reinterpret_cast<const bf16x8*>(&Ks[cur][tok][16 + (hi ? 8 : 0)]);
            S[c] = wmma_bf16(qf[0], kb, S[c]);
            kb.h[0] = *reinterpret_cast<const bf16x8*>(&Ks[cur][tok][32 + (hi ? 8 : 0)]);
            kb.h[1] = *reinterpret_cast<const bf16x8*>(&Ks[cur][tok][48 + (hi ? 8 : 0)]);
            S[c] = wmma_bf16(qf[1], kb, S[c]);
        }
        // + relative position bias (rel_index/bias_table are L2-resident)
#pragma unroll
        for (int c = 0; c < 2; ++c) {
            const int col = ck + c * 16 + l16;
#pragma unroll
            for (int r = 0; r < 8; ++r) {
                const int row = qt * 16 + (hi ? r + 8 : r);
                const int idx = rel_index[row * LTOK + col];
                S[c][r] += bias_table[idx * NH + h];
            }
        }
        // Online softmax: rows 0-7 live in lanes 0-15, rows 8-15 in lanes 16-31
        float scal[8];
#pragma unroll
        for (int r = 0; r < 8; ++r) {
            float mt = fmaxf(S[0][r], S[1][r]);
            mt = fmaxf(mt, __shfl_xor(mt, 8, 16));
            mt = fmaxf(mt, __shfl_xor(mt, 4, 16));
            mt = fmaxf(mt, __shfl_xor(mt, 2, 16));
            mt = fmaxf(mt, __shfl_xor(mt, 1, 16));
            const float mnew = fmaxf(mrow[r], mt);
            scal[r] = __expf(mrow[r] - mnew);
            mrow[r] = mnew;
        }
#pragma unroll
        for (int c = 0; c < 2; ++c)
#pragma unroll
            for (int r = 0; r < 8; ++r)
                S[c][r] = __expf(S[c][r] - mrow[r]);
#pragma unroll
        for (int r = 0; r < 8; ++r) {
            float rs = S[0][r] + S[1][r];
            rs += __shfl_xor(rs, 8, 16);
            rs += __shfl_xor(rs, 4, 16);
            rs += __shfl_xor(rs, 2, 16);
            rs += __shfl_xor(rs, 1, 16);
            lsum[r] = lsum[r] * scal[r] + rs;
        }
#pragma unroll
        for (int nt = 0; nt < 4; ++nt)
#pragma unroll
            for (int r = 0; r < 8; ++r)
                O[nt][r] *= scal[r];
        // C-layout P -> LDS -> A-fragment layout (intra-wave, DS is in-order)
#pragma unroll
        for (int c = 0; c < 2; ++c)
#pragma unroll
            for (int r = 0; r < 8; ++r)
                Pbuf[wave][hi ? r + 8 : r][c * 16 + l16] = f2bf(S[c][r]);
        asm volatile("s_wait_dscnt 0" ::: "memory");
        AFrag pf;
        pf.h[0] = *reinterpret_cast<const bf16x8*>(&Pbuf[wave][l16][hi ? 8 : 0]);
        pf.h[1] = *reinterpret_cast<const bf16x8*>(&Pbuf[wave][l16][16 + (hi ? 8 : 0)]);
        // O += P(16x32) x V(32x16) per 16-wide d tile
#pragma unroll
        for (int nt = 0; nt < 4; ++nt) {
            const int d = nt * 16 + l16;
            AFrag vb;
            vb.h[0] = *reinterpret_cast<const bf16x8*>(&Vs[cur][d][hi ? 8 : 0]);
            vb.h[1] = *reinterpret_cast<const bf16x8*>(&Vs[cur][d][16 + (hi ? 8 : 0)]);
            O[nt] = wmma_bf16(pf, vb, O[nt]);
        }
        __syncthreads();          // readers done: buffer reusable by DMA at i+1
    }

    // Normalize and store bf16 head-slice into [B,L,768]
#pragma unroll
    for (int nt = 0; nt < 4; ++nt)
#pragma unroll
        for (int r = 0; r < 8; ++r) {
            const int rowL = qt * 16 + (hi ? r + 8 : r);
            const int col  = h * HD + nt * 16 + l16;
            xout[((size_t)b * LTOK + rowL) * DIMC + col] = f2bf(O[nt][r] / lsum[r]);
        }
}

// ---------------------------------------------------------------------------
// Kernel 3: output projection [16384,768](bf16) x [768,768](f32->bf16) + b_proj
// A tile is a pure bf16 copy -> double-buffered async global->LDS DMA that
// overlaps the WMMAs; W tile converts f32->bf16 in registers.
// ---------------------------------------------------------------------------
__global__ __launch_bounds__(256)
void proj_gemm_kernel(const __bf16* __restrict__ A,    // [16384,768] bf16
                      const float*  __restrict__ W,    // [768,768]
                      const float*  __restrict__ bvec, // [768]
                      float* __restrict__ out) {       // [16384,768] f32
    __shared__ __bf16 As [2][BM][AP];
    __shared__ __bf16 BsT[BN][BP];

    const int tid  = threadIdx.x;
    const int lane = tid & 31;
    const int wave = tid >> 5;
    const int wm   = wave & 3;
    const int wn   = wave >> 2;
    const int hi   = (lane >= 16);
    const int l16  = lane & 15;

    const int mBlock = blockIdx.y * BM;
    const int nBlock = blockIdx.x * BN;

    // Async A-tile assignment: 256 thr * 32 B = 8 KB tile per K-step
    const int ar = tid >> 1;
    const int ac = (tid & 1) * 16;
    const unsigned aLds0[2] = { lds_addr_of(&As[0][ar][ac]),
                                lds_addr_of(&As[1][ar][ac]) };
    const unsigned aLds1[2] = { lds_addr_of(&As[0][ar][ac + 8]),
                                lds_addr_of(&As[1][ar][ac + 8]) };
    const __bf16* gA = &A[(size_t)(mBlock + ar) * DIMC + ac];

    v8f acc[2][2];
    v8f zero = {};
    acc[0][0] = zero; acc[0][1] = zero; acc[1][0] = zero; acc[1][1] = zero;

    // Pipeline prologue: DMA K-step 0 into buffer 0
    async_copy_b128(aLds0[0], gA);
    async_copy_b128(aLds1[0], gA + 8);

    const int nsteps = DIMC / BK;   // 24
    for (int i = 0; i < nsteps; ++i) {
        const int k0  = i * BK;
        const int cur = i & 1;
        if (i + 1 < nsteps) {   // stream next A tile while computing this one
            const int nxt = cur ^ 1;
            async_copy_b128(aLds0[nxt], gA + k0 + BK);
            async_copy_b128(aLds1[nxt], gA + k0 + BK + 8);
        }
        {   // B tile transposed, f32 -> bf16 (overlaps the DMA)
            const int n4 = (tid & 15) * 4;
            const int kr = tid >> 4;
#pragma unroll
            for (int kk = 0; kk < 2; ++kk) {
                const int k = kr + kk * 16;
                const float4 f = *reinterpret_cast<const float4*>(
                    &W[(size_t)(k0 + k) * DIMC + nBlock + n4]);
                BsT[n4 + 0][k] = f2bf(f.x);
                BsT[n4 + 1][k] = f2bf(f.y);
                BsT[n4 + 2][k] = f2bf(f.z);
                BsT[n4 + 3][k] = f2bf(f.w);
            }
        }
        if (i + 1 < nsteps) wait_asynccnt<2>(); else wait_asynccnt<0>();
        __syncthreads();
#pragma unroll
        for (int mt = 0; mt < 2; ++mt) {
            AFrag a;
            const int row = wm * 32 + mt * 16 + l16;
            a.h[0] = *reinterpret_cast<const bf16x8*>(&As[cur][row][hi ? 8 : 0]);
            a.h[1] = *reinterpret_cast<const bf16x8*>(&As[cur][row][16 + (hi ? 8 : 0)]);
#pragma unroll
            for (int nt = 0; nt < 2; ++nt) {
                AFrag b;
                const int col = wn * 32 + nt * 16 + l16;
                b.h[0] = *reinterpret_cast<const bf16x8*>(&BsT[col][hi ? 8 : 0]);
                b.h[1] = *reinterpret_cast<const bf16x8*>(&BsT[col][16 + (hi ? 8 : 0)]);
                acc[mt][nt] = wmma_bf16(a, b, acc[mt][nt]);
            }
        }
        __syncthreads();
    }
#pragma unroll
    for (int mt = 0; mt < 2; ++mt)
#pragma unroll
        for (int nt = 0; nt < 2; ++nt)
#pragma unroll
            for (int r = 0; r < 8; ++r) {
                const int gr = mBlock + wm * 32 + mt * 16 + (hi ? r + 8 : r);
                const int gc = nBlock + wn * 32 + nt * 16 + l16;
                out[(size_t)gr * DIMC + gc] = acc[mt][nt][r] + bvec[gc];
            }
}

// ---------------------------------------------------------------------------
// Host launcher.  Workspace: 4 x 25,165,824 B bf16 buffers (~100.7 MB).
// ---------------------------------------------------------------------------
extern "C" void kernel_launch(void* const* d_in, const int* in_sizes, int n_in,
                              void* d_out, int out_size, void* d_ws, size_t ws_size,
                              hipStream_t stream) {
    (void)in_sizes; (void)n_in; (void)out_size; (void)ws_size;
    const float* qkv        = (const float*)d_in[0];
    const float* W_qkv      = (const float*)d_in[1];
    const float* b_qkv      = (const float*)d_in[2];
    const float* bias_table = (const float*)d_in[3];
    const float* W_proj     = (const float*)d_in[4];
    const float* b_proj     = (const float*)d_in[5];
    const int*   rel_index  = (const int*)d_in[6];

    char* ws = (char*)d_ws;
    const size_t per = (size_t)BSZ * NH * LTOK * HD * sizeof(__bf16); // 25,165,824
    __bf16* q_ws  = (__bf16*)(ws);
    __bf16* k_ws  = (__bf16*)(ws + per);
    __bf16* vt_ws = (__bf16*)(ws + 2 * per);
    __bf16* x_ws  = (__bf16*)(ws + 3 * per);

    qkv_gemm_kernel<<<dim3(NQKV / BN, MTOT / BM), 256, 0, stream>>>(
        qkv, W_qkv, b_qkv, q_ws, k_ws, vt_ws);

    const int totalWaves = BSZ * NH * (LTOK / 16);   // 12288
    attn_kernel<<<dim3(totalWaves / 4), 128, 0, stream>>>(
        q_ws, k_ws, vt_ws, bias_table, rel_index, x_ws);

    proj_gemm_kernel<<<dim3(DIMC / BN, MTOT / BM), 256, 0, stream>>>(
        x_ws, W_proj, b_proj, (float*)d_out);
}
